// Sequence_55422257988009
// MI455X (gfx1250) — compile-verified, matrix operand order checked by reading, not run on hardware
//
#include <hip/hip_runtime.h>
#include <hip/hip_bf16.h>

// ---------------------------------------------------------------------------
// 2-layer LSTM (H=51), B=2048, T=256 (+16 future) for MI455X (gfx1250).
//
// Latency-bound recurrence (~35 GFLOP total, ~4 MB HBM traffic) -> keep all
// state on-chip and minimize the per-timestep critical path.
//
//  * 128 workgroups x 4 wave32 (one per SIMD of the WGP). Each workgroup owns
//    16 batch rows; the 13 gate N-tiles are split {4,3,3,3} across the waves
//    so all four matrix pipes of the WGP cooperate on each timestep.
//  * WMMA B-operands are TIME-INVARIANT -> each wave gathers its 3 tiles'
//    operands (18 x v16h = 144 VGPRs) straight from global memory into the
//    documented B layout ONCE and keeps them in registers for all 272 steps:
//    no per-step B loads, no s_wait_dscnt in front of the WMMAs.
//    Wave0's 4th tile (#12) uses a small LDS copy to cap register pressure.
//  * GEMMs via v_wmma_f32_16x16x32_f16:
//      layer1: gates1 = x*wih1 + bsum1 + h1 @ Whh1^T      (2 WMMA / tile)
//      layer2: gates2 = bsum2 + h1@Wih2^T + h2@Whh2^T     (4 WMMA / tile)
//  * h-state f16 [16][64] in LDS -> A operand = two ds_load_b128 per K-chunk.
//  * activations on native trans units (v_tanh_f32); sigmoid = 0.5*tanh+0.5.
//  * x[t+1] prefetched into registers during step t; output projection:
//    8 lanes/row + __shfl_xor reduce; xv hand-off folded into the final
//    barrier of the step (5 barriers/step).
// ---------------------------------------------------------------------------

typedef __attribute__((ext_vector_type(16))) _Float16 v16h;
typedef __attribute__((ext_vector_type(8)))  _Float16 v8h;
typedef __attribute__((ext_vector_type(8)))  float    v8f;

#define HSZ 51          // hidden size
#define G4  204         // 4*H
#define NRT 13          // real N tiles (13*16 = 208 >= 204)
#define NP  208         // gate scratch width
#define KP  64          // padded K (two 32-chunks)
#define NWAVES 4
#define NTHREADS (NWAVES * 32)

struct SM {
  alignas(32) _Float16 wB12[6 * 512];   // tile #12 B-operands (3 mats x 2 kt)
  alignas(16) _Float16 h1[16 * KP];     // layer-1 hidden (f16, zero-padded)
  alignas(16) _Float16 h2[16 * KP];     // layer-2 hidden
  alignas(16) float xv[16];             // current per-row input x_t
  float gates[16 * NP];                 // per-step gate scratch (f32)
  float c1[16 * 52];                    // cell states (f32)
  float c2[16 * 52];
};

__device__ __forceinline__ float fast_tanh(float x) {
#if __has_builtin(__builtin_amdgcn_tanhf)
  return __builtin_amdgcn_tanhf(x);     // native V_TANH_F32 trans op
#else
  return tanhf(x);
#endif
}
__device__ __forceinline__ float sigm(float x) {
  return fmaf(0.5f, fast_tanh(0.5f * x), 0.5f);
}

// A-operand (16x32 f16) per documented layout:
//   lane L: M = L&15;  halves 0-7  -> K = kt*32 + (L>>4)*8 + 0..7
//                      halves 8-15 -> K = kt*32 + (L>>4)*8 + 16..23
__device__ __forceinline__ v16h load_A(const _Float16* hb, int lane, int kt) {
  const int M = lane & 15, hi = lane >> 4;
  const _Float16* p = hb + M * KP + kt * 32 + hi * 8;
  v8h lo = *(const v8h*)(p);
  v8h hh = *(const v8h*)(p + 16);
  return __builtin_shufflevector(lo, hh, 0, 1, 2, 3, 4, 5, 6, 7,
                                         8, 9, 10, 11, 12, 13, 14, 15);
}

// Gather one B-operand tile (16x16x32 chunk) from global f32 weights into the
// documented B layout (mirror of A with N = lane&15). One-time cost.
__device__ __forceinline__ v16h gatherB(const float* __restrict__ W,
                                        int lane, int nt, int kt) {
  const int N  = nt * 16 + (lane & 15);
  const int kb = kt * 32 + ((lane >> 4) << 3);
  v16h b;
#pragma unroll
  for (int h = 0; h < 16; ++h) {
    const int K = ((h >> 3) << 4) + kb + (h & 7);
    const float v = (N < G4 && K < HSZ) ? W[N * HSZ + K] : 0.f;
    b[h] = (_Float16)v;
  }
  return b;
}

__device__ __forceinline__ v16h ldsB12(const _Float16* base, int lane,
                                       int m, int kt) {
  return *(const v16h*)(base + ((m * 2 + kt) << 9) + lane * 16);
}

__global__ void __launch_bounds__(NTHREADS)
lstm_seq_kernel(const float* __restrict__ inp,
                const float* __restrict__ Wih1, const float* __restrict__ bih1,
                const float* __restrict__ Whh1, const float* __restrict__ bhh1,
                const float* __restrict__ Wih2, const float* __restrict__ bih2,
                const float* __restrict__ Whh2, const float* __restrict__ bhh2,
                const float* __restrict__ Wlin, const float* __restrict__ pblin,
                float* __restrict__ out, int T, int TF)
{
  extern __shared__ char smraw[];
  SM& s = *reinterpret_cast<SM*>(smraw);

  const int tid  = threadIdx.x;     // 0..127
  const int lane = tid & 31;
  const int w    = tid >> 5;        // wave id 0..3
  const int hi   = lane >> 4;
  const int ml   = lane & 15;
  const int b0   = blockIdx.x * 16;

  // ---- zero-init on-chip state -----------------------------------------
  for (int p = tid; p < 16 * KP; p += NTHREADS) {
    s.h1[p] = (_Float16)0.f;
    s.h2[p] = (_Float16)0.f;
  }
  for (int p = tid; p < 16 * 52; p += NTHREADS) { s.c1[p] = 0.f; s.c2[p] = 0.f; }
  if (tid < 16) s.xv[tid] = (T > 0) ? inp[(b0 + tid) * T] : 0.f;

  // ---- tile #12 B-operands into LDS (used by wave0 only) ----------------
  {
    const float* mats[3] = { Whh1, Wih2, Whh2 };
    for (int idx = tid; idx < 6 * 512; idx += NTHREADS) {
      const int chunk = idx >> 9;          // 0..5 -> (m, kt)
      const int m  = chunk >> 1;
      const int kt = chunk & 1;
      const int within = idx & 511;
      const int lp = within >> 4;
      const int h  = within & 15;
      const int K = kt * 32 + ((h >> 3) << 4) + ((lp >> 4) << 3) + (h & 7);
      const int N = 192 + (lp & 15);
      const float v = (N < G4 && K < HSZ) ? mats[m][N * HSZ + K] : 0.f;
      s.wB12[idx] = (_Float16)v;
    }
  }

  // ---- per-wave register-resident weights (tiles nt = 4q + w, q<3) ------
  v16h B1[3][2], B2i[3][2], B2h[3][2];
  float wv1[3], bv1[3], bv2[3];
#pragma unroll
  for (int q = 0; q < 3; ++q) {
    const int nt = q * 4 + w;            // 0..11, always < 13
    B1 [q][0] = gatherB(Whh1, lane, nt, 0);
    B1 [q][1] = gatherB(Whh1, lane, nt, 1);
    B2i[q][0] = gatherB(Wih2, lane, nt, 0);
    B2i[q][1] = gatherB(Wih2, lane, nt, 1);
    B2h[q][0] = gatherB(Whh2, lane, nt, 0);
    B2h[q][1] = gatherB(Whh2, lane, nt, 1);
    const int rb = nt * 16 + ml;         // < 192+16 <= 207
    const bool ok = (rb < G4);
    wv1[q] = ok ? Wih1[rb] : 0.f;
    bv1[q] = ok ? (bih1[rb] + bhh1[rb]) : 0.f;
    bv2[q] = ok ? (bih2[rb] + bhh2[rb]) : 0.f;
  }
  // wave0's 4th tile (#12) scalars
  float wv12 = 0.f, bv12a = 0.f, bv12b = 0.f;
  if (w == 0) {
    const int rb = 192 + ml;
    const bool ok = (rb < G4);
    wv12  = ok ? Wih1[rb] : 0.f;
    bv12a = ok ? (bih1[rb] + bhh1[rb]) : 0.f;
    bv12b = ok ? (bih2[rb] + bhh2[rb]) : 0.f;
  }
  // projection weights: 8 lanes per row, 8 cols each (zero padded)
  float wl[8];
  {
    const int sub = tid & 7;
#pragma unroll
    for (int j = 0; j < 8; ++j) {
      const int col = sub * 8 + j;
      wl[j] = (col < HSZ) ? Wlin[col] : 0.f;
    }
  }
  const float bl = pblin[0];
  __syncthreads();

  float xnext = 0.f;

  // ---- time loop --------------------------------------------------------
  for (int t = 0; t < TF; ++t) {
    // x_t is in s.xv (written before the barrier that ended the last step)
    const float4 xa = *(const float4*)&s.xv[hi * 8];
    const float4 xb = *(const float4*)&s.xv[hi * 8 + 4];
    const float xr[8] = { xa.x, xa.y, xa.z, xa.w, xb.x, xb.y, xb.z, xb.w };

    // prefetch next timestep's input (L2-resident) under the GEMMs
    if (tid < 16 && t + 1 < T) xnext = inp[(b0 + tid) * T + t + 1];

    // ---- layer 1: gates1 = x*wih1 + bsum1 + h1 @ Whh1^T ----
    const v16h a0 = load_A(s.h1, lane, 0);
    const v16h a1 = load_A(s.h1, lane, 1);
#pragma unroll
    for (int q = 0; q < 3; ++q) {
      const int rbase = (q * 4 + w) * 16 + ml;
      v8f acc;
#pragma unroll
      for (int v = 0; v < 8; ++v) acc[v] = fmaf(xr[v], wv1[q], bv1[q]);
      acc = __builtin_amdgcn_wmma_f32_16x16x32_f16(
          false, a0, false, B1[q][0], (short)0, acc, false, false);
      acc = __builtin_amdgcn_wmma_f32_16x16x32_f16(
          false, a1, false, B1[q][1], (short)0, acc, false, false);
#pragma unroll
      for (int v = 0; v < 8; ++v)
        s.gates[(v + 8 * hi) * NP + rbase] = acc[v];
    }
    if (w == 0) {                        // tile #12 (B from LDS)
      const int rbase = 192 + ml;
      v8f acc;
#pragma unroll
      for (int v = 0; v < 8; ++v) acc[v] = fmaf(xr[v], wv12, bv12a);
      acc = __builtin_amdgcn_wmma_f32_16x16x32_f16(
          false, a0, false, ldsB12(s.wB12, lane, 0, 0), (short)0, acc, false, false);
      acc = __builtin_amdgcn_wmma_f32_16x16x32_f16(
          false, a1, false, ldsB12(s.wB12, lane, 0, 1), (short)0, acc, false, false);
#pragma unroll
      for (int v = 0; v < 8; ++v)
        s.gates[(v + 8 * hi) * NP + rbase] = acc[v];
    }
    __syncthreads();

    // layer-1 elementwise (PyTorch gate order i,f,g,o)
    for (int e = tid; e < 16 * HSZ; e += NTHREADS) {
      const int j = e >> 4, row = e & 15;
      const float gi = s.gates[row * NP + j];
      const float gf = s.gates[row * NP + HSZ + j];
      const float gg = s.gates[row * NP + 2 * HSZ + j];
      const float go = s.gates[row * NP + 3 * HSZ + j];
      const float c = sigm(gf) * s.c1[row * 52 + j] + sigm(gi) * fast_tanh(gg);
      s.c1[row * 52 + j] = c;
      s.h1[row * KP + j] = (_Float16)(sigm(go) * fast_tanh(c));
    }
    __syncthreads();

    // ---- layer 2: gates2 = bsum2 + h1_new @ Wih2^T + h2 @ Whh2^T ----
    const v16h p0 = load_A(s.h1, lane, 0);
    const v16h p1 = load_A(s.h1, lane, 1);
    const v16h q0 = load_A(s.h2, lane, 0);
    const v16h q1 = load_A(s.h2, lane, 1);
#pragma unroll
    for (int q = 0; q < 3; ++q) {
      const int rbase = (q * 4 + w) * 16 + ml;
      v8f acc;
#pragma unroll
      for (int v = 0; v < 8; ++v) acc[v] = bv2[q];
      acc = __builtin_amdgcn_wmma_f32_16x16x32_f16(
          false, p0, false, B2i[q][0], (short)0, acc, false, false);
      acc = __builtin_amdgcn_wmma_f32_16x16x32_f16(
          false, p1, false, B2i[q][1], (short)0, acc, false, false);
      acc = __builtin_amdgcn_wmma_f32_16x16x32_f16(
          false, q0, false, B2h[q][0], (short)0, acc, false, false);
      acc = __builtin_amdgcn_wmma_f32_16x16x32_f16(
          false, q1, false, B2h[q][1], (short)0, acc, false, false);
#pragma unroll
      for (int v = 0; v < 8; ++v)
        s.gates[(v + 8 * hi) * NP + rbase] = acc[v];
    }
    if (w == 0) {                        // tile #12 (B from LDS)
      const int rbase = 192 + ml;
      v8f acc;
#pragma unroll
      for (int v = 0; v < 8; ++v) acc[v] = bv12b;
      acc = __builtin_amdgcn_wmma_f32_16x16x32_f16(
          false, p0, false, ldsB12(s.wB12, lane, 1, 0), (short)0, acc, false, false);
      acc = __builtin_amdgcn_wmma_f32_16x16x32_f16(
          false, p1, false, ldsB12(s.wB12, lane, 1, 1), (short)0, acc, false, false);
      acc = __builtin_amdgcn_wmma_f32_16x16x32_f16(
          false, q0, false, ldsB12(s.wB12, lane, 2, 0), (short)0, acc, false, false);
      acc = __builtin_amdgcn_wmma_f32_16x16x32_f16(
          false, q1, false, ldsB12(s.wB12, lane, 2, 1), (short)0, acc, false, false);
#pragma unroll
      for (int v = 0; v < 8; ++v)
        s.gates[(v + 8 * hi) * NP + rbase] = acc[v];
    }
    __syncthreads();

    for (int e = tid; e < 16 * HSZ; e += NTHREADS) {
      const int j = e >> 4, row = e & 15;
      const float gi = s.gates[row * NP + j];
      const float gf = s.gates[row * NP + HSZ + j];
      const float gg = s.gates[row * NP + 2 * HSZ + j];
      const float go = s.gates[row * NP + 3 * HSZ + j];
      const float c = sigm(gf) * s.c2[row * 52 + j] + sigm(gi) * fast_tanh(gg);
      s.c2[row * 52 + j] = c;
      s.h2[row * KP + j] = (_Float16)(sigm(go) * fast_tanh(c));
    }
    __syncthreads();

    // ---- output projection + x hand-off for t+1 ----
    {
      const int row = tid >> 3;          // 0..15
      const int sub = tid & 7;           // 0..7
      float o = 0.f;
#pragma unroll
      for (int j = 0; j < 8; ++j)
        o = fmaf((float)s.h2[row * KP + sub * 8 + j], wl[j], o);
      o += __shfl_xor(o, 1, 32);
      o += __shfl_xor(o, 2, 32);
      o += __shfl_xor(o, 4, 32);
      if (sub == 0) {
        o += bl;
        out[(b0 + row) * TF + t] = o;
        if (t + 1 >= T) s.xv[row] = o;   // autoregressive feedback
      }
      if (tid < 16 && t + 1 < T) s.xv[tid] = xnext;  // disjoint (uniform in t)
    }
    __syncthreads();
  }
}

extern "C" void kernel_launch(void* const* d_in, const int* in_sizes, int n_in,
                              void* d_out, int out_size, void* d_ws, size_t ws_size,
                              hipStream_t stream) {
  const float* inputs = (const float*)d_in[0];
  const float* Wih1   = (const float*)d_in[1];
  const float* bih1   = (const float*)d_in[2];
  const float* Whh1   = (const float*)d_in[3];
  const float* bhh1   = (const float*)d_in[4];
  const float* Wih2   = (const float*)d_in[5];
  const float* bih2   = (const float*)d_in[6];
  const float* Whh2   = (const float*)d_in[7];
  const float* bhh2   = (const float*)d_in[8];
  const float* Wlin   = (const float*)d_in[9];
  const float* blin   = (const float*)d_in[10];
  float* out = (float*)d_out;

  const int B  = 2048;
  const int T  = in_sizes[0] / B;   // 256
  const int TF = out_size / B;      // 272 (= T + future)

  dim3 grid(B / 16), block(NTHREADS);
  size_t shbytes = sizeof(SM);      // ~30 KB dynamic LDS
  lstm_seq_kernel<<<grid, block, shbytes, stream>>>(
      inputs, Wih1, bih1, Whh1, bhh1, Wih2, bih2, Whh2, bhh2,
      Wlin, blin, out, T, TF);
}
